// DynamicResBlock_45552423141978
// MI455X (gfx1250) — matrix operand order repaired and checked by reading, not compiled, not run on hardware
//
#include <hip/hip_runtime.h>
#include <hip/hip_bf16.h>
#include <math.h>

typedef __attribute__((ext_vector_type(16))) _Float16 v16h;
typedef __attribute__((ext_vector_type(8)))  float    v8f;

#define B_     8
#define C_     128
#define H_     192
#define W_     192
#define A_     16
#define KSTEPS 36            // K = 9*128 = 1152, 32 per step
#define TILE_W 64
#define LDS_PC 68            // cip dimension padded 64 -> 68 (bank rotate by 4/px)
#define NPX    66            // 64 + 2 halo pixels staged per row
#define WDW    73728         // packed-weight dwords per conv (36*8*32*8)

// workspace byte offsets
#define OFF_P    0
#define OFF_CH   4096
#define OFF_FL   8192
#define OFF_SP   12288
#define OFF_W1   12800                         // 8 per-batch copies, ch+sp folded
#define OFF_W2   (OFF_W1 + 8 * WDW * 4)
#define OFF_Y    (OFF_W2 + WDW * 4)            // pair-interleaved f16 y, 8*64*H*W dwords

// ---------------- global average pool: p[b][c] = mean(x[b,c,:,:]) -------------
__global__ void pool_kernel(const float* __restrict__ x, float* __restrict__ p) {
    int bc = blockIdx.x;
    const float* src = x + (size_t)bc * (H_ * W_);
    float s = 0.f;
    for (int i = threadIdx.x; i < H_ * W_; i += 256) s += src[i];
    __shared__ float red[256];
    red[threadIdx.x] = s;
    __syncthreads();
    for (int off = 128; off > 0; off >>= 1) {
        if ((int)threadIdx.x < off) red[threadIdx.x] += red[threadIdx.x + off];
        __syncthreads();
    }
    if (threadIdx.x == 0) p[bc] = red[0] * (1.0f / (H_ * W_));
}

// ---------------- attention head: fc+bn+relu, then ch / fl / sp sigmoids ------
__global__ void att_kernel(const float* __restrict__ p,
                           const float* __restrict__ fc_w,
                           const float* __restrict__ bn_g, const float* __restrict__ bn_b,
                           const float* __restrict__ bn_m, const float* __restrict__ bn_v,
                           const float* __restrict__ ch_w, const float* __restrict__ ch_b,
                           const float* __restrict__ fl_w, const float* __restrict__ fl_b,
                           const float* __restrict__ sp_w, const float* __restrict__ sp_b,
                           float* __restrict__ ch_o, float* __restrict__ fl_o,
                           float* __restrict__ sp_o) {
    __shared__ float pl[B_ * C_];
    __shared__ float yy[B_ * A_];
    for (int i = threadIdx.x; i < B_ * C_; i += 256) pl[i] = p[i];
    __syncthreads();
    if (threadIdx.x < B_ * A_) {
        int b = threadIdx.x >> 4, a = threadIdx.x & 15;
        float s = 0.f;
        for (int c = 0; c < C_; ++c) s += pl[b * C_ + c] * fc_w[a * C_ + c];
        s = (s - bn_m[a]) * rsqrtf(bn_v[a] + 1e-5f) * bn_g[a] + bn_b[a];
        yy[threadIdx.x] = fmaxf(s, 0.f);
    }
    __syncthreads();
    for (int i = threadIdx.x; i < B_ * C_; i += 256) {
        int b = i >> 7, c = i & 127;
        float s1 = ch_b[c], s2 = fl_b[c];
        for (int a = 0; a < A_; ++a) {
            float yv = yy[b * A_ + a];
            s1 += yv * ch_w[c * A_ + a];
            s2 += yv * fl_w[c * A_ + a];
        }
        ch_o[i] = 1.f / (1.f + expf(-s1));
        fl_o[i] = 1.f / (1.f + expf(-s2));
    }
    if (threadIdx.x < B_ * 9) {
        int b = threadIdx.x / 9, j = threadIdx.x % 9;
        float s = sp_b[j];
        for (int a = 0; a < A_; ++a) s += yy[b * A_ + a] * sp_w[j * A_ + a];
        sp_o[threadIdx.x] = 1.f / (1.f + expf(-s));
    }
}

// A(16x32) f16 fragment layout (per ISA):
//   lanes 0-15:  M=lane,    K 0-7  in V0-3, K 16-23 in V4-7
//   lanes 16-31: M=lane-16, K 8-15 in V0-3, K 24-31 in V4-7
// K ordering: k = (kh*3+kw)*128 + ci.  dword index d = ((ks*8+mt)*32+lane)*8+v

// pack conv1 weights per batch with ch & sp attention folded in
__global__ void pack1_kernel(const float* __restrict__ w,
                             const float* __restrict__ ch, const float* __restrict__ sp,
                             unsigned int* __restrict__ dst) {
    int flat = blockIdx.x * 256 + threadIdx.x;        // 8 * 73728
    if (flat >= B_ * WDW) return;
    int b = flat / WDW;
    int d = flat - b * WDW;
    int v = d & 7, lane = (d >> 3) & 31, mt = (d >> 8) & 7, ks = d >> 11;
    int m = mt * 16 + (lane & 15);
    unsigned int out = 0;
    for (int j = 0; j < 2; ++j) {
        int kl = (v < 4 ? 2 * v + j : 16 + 2 * (v - 4) + j) + ((lane >= 16) ? 8 : 0);
        int kg = ks * 32 + kl;
        int khw = kg >> 7, ci = kg & 127;
        int kh = khw / 3, kw = khw % 3;
        float wv = w[(((size_t)m * C_ + ci) * 3 + kh) * 3 + kw]
                 * ch[b * C_ + ci] * sp[b * 9 + khw];
        union { _Float16 h; unsigned short u; } cv;
        cv.h = (_Float16)wv;
        out |= ((unsigned int)cv.u) << (16 * j);
    }
    dst[flat] = out;
}

// pack conv2 weights (shared across batch)
__global__ void pack2_kernel(const float* __restrict__ w, unsigned int* __restrict__ dst) {
    int d = blockIdx.x * 256 + threadIdx.x;
    if (d >= WDW) return;
    int v = d & 7, lane = (d >> 3) & 31, mt = (d >> 8) & 7, ks = d >> 11;
    int m = mt * 16 + (lane & 15);
    unsigned int out = 0;
    for (int j = 0; j < 2; ++j) {
        int kl = (v < 4 ? 2 * v + j : 16 + 2 * (v - 4) + j) + ((lane >= 16) ? 8 : 0);
        int kg = ks * 32 + kl;
        int khw = kg >> 7, ci = kg & 127;
        int kh = khw / 3, kw = khw % 3;
        float wv = w[(((size_t)m * C_ + ci) * 3 + kh) * 3 + kw];
        union { _Float16 h; unsigned short u; } cv;
        cv.h = (_Float16)wv;
        out |= ((unsigned int)cv.u) << (16 * j);
    }
    dst[d] = out;
}

union AF  { v16h h; uint4 u4[2]; };
union BF  { v16h h; uint4 u4[2]; };
union PK  { _Float16 h[2]; unsigned int u; };
union PK2 { _Float16 h[4]; uint2 u2; };

// ------------- conv1: y = fl * dynconv(x, w1_folded), pair-interleaved out ----
__global__ __launch_bounds__(256, 2)
void conv1_kernel(const float* __restrict__ x, const unsigned int* __restrict__ wall,
                  const float* __restrict__ fl, unsigned int* __restrict__ yq) {
    // LDS: [row 0..2][px 0..65][cip 0..63 pad 68], dword = {ci even, ci odd} f16
    __shared__ unsigned int lds[3 * NPX * LDS_PC];
    int bx  = blockIdx.x;
    int b   = bx / (H_ * 3);
    int rem = bx % (H_ * 3);
    int h   = rem / 3;
    int px0 = (rem % 3) * TILE_W;
    int tid = threadIdx.x;
    const unsigned int* wpk = wall + (size_t)b * WDW;

    // fill: each task = one pixel x one channel-quad (2 dwords, ds_store_b64)
    for (int i = tid; i < 3 * NPX * 32; i += 256) {
        int px = i % NPX;
        int rc = i / NPX;
        int q  = rc & 31;          // channel quad: channels 4q..4q+3
        int r  = rc >> 5;
        int gh = h - 1 + r;
        int gw = px0 - 1 + px;
        PK2 pk; pk.u2.x = 0; pk.u2.y = 0;
        if ((unsigned)gh < (unsigned)H_ && (unsigned)gw < (unsigned)W_) {
            const float* xp = &x[(((size_t)b * C_ + 4 * q) * H_ + gh) * W_ + gw];
            pk.h[0] = (_Float16)xp[0];
            pk.h[1] = (_Float16)xp[(size_t)1 * H_ * W_];
            pk.h[2] = (_Float16)xp[(size_t)2 * H_ * W_];
            pk.h[3] = (_Float16)xp[(size_t)3 * H_ * W_];
        }
        *(uint2*)&lds[(r * NPX + px) * LDS_PC + 2 * q] = pk.u2;
    }
    __syncthreads();

    int lane = tid & 31;
    int wv_i = (tid >> 5) & 3;        // M-tile pair index
    int wv_j = tid >> 7;              // N-tile pair index
    int n    = lane & 15;
    int kbp  = (lane >> 4) * 4;       // (K base)/2 per lane half
    v8f acc[2][2] = {{v8f{}, v8f{}}, {v8f{}, v8f{}}};

    for (int ks = 0; ks < KSTEPS; ++ks) {
        int khw  = ks >> 2;
        int cibp = (ks & 3) * 16;     // (ci base)/2
        int kh = khw / 3, kw = khw % 3;

        AF a0, a1;
        const uint4* wp = (const uint4*)wpk + ((size_t)(ks * 8 + 2 * wv_i) * 32 + lane) * 2;
        a0.u4[0] = wp[0];  a0.u4[1] = wp[1];
        a1.u4[0] = wp[64]; a1.u4[1] = wp[65];     // next M-tile: +32 lanes * 2 uint4

        #pragma unroll
        for (int t = 0; t < 2; ++t) {
            int nt = 2 * wv_j + t;
            int px = nt * 16 + n + kw;
            // two 4-dword runs (K 8h..8h+7, 16+8h..), 16B aligned -> two ds_load_b128
            const uint4* bp = (const uint4*)&lds[(kh * NPX + px) * LDS_PC + cibp + kbp];
            BF bf;
            bf.u4[0] = bp[0];
            bf.u4[1] = bp[2];
            acc[t][0] = __builtin_amdgcn_wmma_f32_16x16x32_f16(
                false, a0.h, false, bf.h, (short)0, acc[t][0], false, false);
            acc[t][1] = __builtin_amdgcn_wmma_f32_16x16x32_f16(
                false, a1.h, false, bf.h, (short)0, acc[t][1], false, false);
        }
    }

    int mrow = (lane >> 4) * 8;
    #pragma unroll
    for (int t = 0; t < 2; ++t) {
        int pixel = px0 + (2 * wv_j + t) * 16 + n;
        #pragma unroll
        for (int m = 0; m < 2; ++m) {
            int co0 = (2 * wv_i + m) * 16 + mrow;
            #pragma unroll
            for (int vi = 0; vi < 8; vi += 2) {
                PK pk;
                pk.h[0] = (_Float16)(acc[t][m][vi]     * fl[b * C_ + co0 + vi]);
                pk.h[1] = (_Float16)(acc[t][m][vi + 1] * fl[b * C_ + co0 + vi + 1]);
                int cop = (co0 + vi) >> 1;
                yq[(((size_t)b * 64 + cop) * H_ + h) * W_ + pixel] = pk.u;
            }
        }
    }
}

// ------------- conv2: out = x + gelu(conv(y, w2) + b2), f32 out ---------------
__global__ __launch_bounds__(256, 2)
void conv2_kernel(const unsigned int* __restrict__ yq, const unsigned int* __restrict__ wpk,
                  const float* __restrict__ bias, const float* __restrict__ x,
                  float* __restrict__ out) {
    __shared__ unsigned int lds[3 * NPX * LDS_PC];
    int bx  = blockIdx.x;
    int b   = bx / (H_ * 3);
    int rem = bx % (H_ * 3);
    int h   = rem / 3;
    int px0 = (rem % 3) * TILE_W;
    int tid = threadIdx.x;

    for (int i = tid; i < 3 * NPX * 32; i += 256) {
        int px = i % NPX;
        int rc = i / NPX;
        int q  = rc & 31;
        int r  = rc >> 5;
        int gh = h - 1 + r;
        int gw = px0 - 1 + px;
        uint2 v; v.x = 0; v.y = 0;
        if ((unsigned)gh < (unsigned)H_ && (unsigned)gw < (unsigned)W_) {
            const unsigned int* yp = &yq[(((size_t)b * 64 + 2 * q) * H_ + gh) * W_ + gw];
            v.x = yp[0];
            v.y = yp[(size_t)H_ * W_];
        }
        *(uint2*)&lds[(r * NPX + px) * LDS_PC + 2 * q] = v;
    }
    __syncthreads();

    int lane = tid & 31;
    int wv_i = (tid >> 5) & 3;
    int wv_j = tid >> 7;
    int n    = lane & 15;
    int kbp  = (lane >> 4) * 4;
    v8f acc[2][2] = {{v8f{}, v8f{}}, {v8f{}, v8f{}}};

    for (int ks = 0; ks < KSTEPS; ++ks) {
        int khw  = ks >> 2;
        int cibp = (ks & 3) * 16;
        int kh = khw / 3, kw = khw % 3;

        AF a0, a1;
        const uint4* wp = (const uint4*)wpk + ((size_t)(ks * 8 + 2 * wv_i) * 32 + lane) * 2;
        a0.u4[0] = wp[0];  a0.u4[1] = wp[1];
        a1.u4[0] = wp[64]; a1.u4[1] = wp[65];

        #pragma unroll
        for (int t = 0; t < 2; ++t) {
            int nt = 2 * wv_j + t;
            int px = nt * 16 + n + kw;
            const uint4* bp = (const uint4*)&lds[(kh * NPX + px) * LDS_PC + cibp + kbp];
            BF bf;
            bf.u4[0] = bp[0];
            bf.u4[1] = bp[2];
            acc[t][0] = __builtin_amdgcn_wmma_f32_16x16x32_f16(
                false, a0.h, false, bf.h, (short)0, acc[t][0], false, false);
            acc[t][1] = __builtin_amdgcn_wmma_f32_16x16x32_f16(
                false, a1.h, false, bf.h, (short)0, acc[t][1], false, false);
        }
    }

    int mrow = (lane >> 4) * 8;
    #pragma unroll
    for (int t = 0; t < 2; ++t) {
        int pixel = px0 + (2 * wv_j + t) * 16 + n;
        #pragma unroll
        for (int m = 0; m < 2; ++m) {
            #pragma unroll
            for (int vi = 0; vi < 8; ++vi) {
                int co = (2 * wv_i + m) * 16 + mrow + vi;
                float v = acc[t][m][vi] + bias[co];
                float g = 0.5f * v * (1.f + erff(v * 0.70710678118654752f));
                size_t idx = (((size_t)b * C_ + co) * H_ + h) * W_ + pixel;
                out[idx] = x[idx] + g;
            }
        }
    }
}

extern "C" void kernel_launch(void* const* d_in, const int* in_sizes, int n_in,
                              void* d_out, int out_size, void* d_ws, size_t ws_size,
                              hipStream_t stream) {
    (void)in_sizes; (void)n_in; (void)out_size; (void)ws_size;
    const float* x         = (const float*)d_in[0];
    const float* od_weight = (const float*)d_in[1];
    const float* fc_w      = (const float*)d_in[2];
    const float* bn_g      = (const float*)d_in[3];
    const float* bn_b      = (const float*)d_in[4];
    const float* bn_m      = (const float*)d_in[5];
    const float* bn_v      = (const float*)d_in[6];
    const float* ch_w      = (const float*)d_in[7];
    const float* ch_b      = (const float*)d_in[8];
    const float* fl_w      = (const float*)d_in[9];
    const float* fl_b      = (const float*)d_in[10];
    const float* sp_w      = (const float*)d_in[11];
    const float* sp_b      = (const float*)d_in[12];
    const float* conv2_w   = (const float*)d_in[13];
    const float* conv2_b   = (const float*)d_in[14];

    char* ws = (char*)d_ws;
    float*        p_ws  = (float*)(ws + OFF_P);
    float*        ch_ws = (float*)(ws + OFF_CH);
    float*        fl_ws = (float*)(ws + OFF_FL);
    float*        sp_ws = (float*)(ws + OFF_SP);
    unsigned int* w1p   = (unsigned int*)(ws + OFF_W1);
    unsigned int* w2p   = (unsigned int*)(ws + OFF_W2);
    unsigned int* y_ws  = (unsigned int*)(ws + OFF_Y);

    pool_kernel<<<B_ * C_, 256, 0, stream>>>(x, p_ws);
    att_kernel<<<1, 256, 0, stream>>>(p_ws, fc_w, bn_g, bn_b, bn_m, bn_v,
                                      ch_w, ch_b, fl_w, fl_b, sp_w, sp_b,
                                      ch_ws, fl_ws, sp_ws);
    pack1_kernel<<<(B_ * WDW + 255) / 256, 256, 0, stream>>>(od_weight, ch_ws, sp_ws, w1p);
    pack2_kernel<<<(WDW + 255) / 256, 256, 0, stream>>>(conv2_w, w2p);

    int conv_blocks = B_ * H_ * 3;   // 64-pixel row strips
    conv1_kernel<<<conv_blocks, 256, 0, stream>>>(x, w1p, fl_ws, y_ws);
    conv2_kernel<<<conv_blocks, 256, 0, stream>>>(y_ws, w2p, conv2_b, x, (float*)d_out);
}
